// SparseCoding_41807211659987
// MI455X (gfx1250) — matrix operand order repaired and checked by reading, not compiled, not run on hardware
//
#include <hip/hip_runtime.h>

// ---------------------------------------------------------------------------
// Sparse coding (10-step RMSprop+momentum) fused kernel for gfx1250 (MI455X)
//   inputs : [8192, 512]  f32, W : [1024, 512] f32, out : [8192, 1024] f32
//
// Transposed formulation: each workgroup owns 16 batch rows == ONE 16-wide
// N-tile.  Both GEMMs run with code/input dims as M:
//   GEMM1': E^T [512 x 16]  = W^T [512x1024] * x^T [1024x16]   (- inputs^T)
//   GEMM2': G^T [1024 x 16] = W   [1024x512] * E^T [512x16]
// Every WMMA D-fragment holds, per lane, one fixed batch row and 8
// CONSECUTIVE code/input indices -> all LDS traffic is b128-vectorized and
// the RMSprop state a1/a2 (lane-private) lives entirely in VGPRs.
// W / W^T are packed once into A-fragment tiles (32 B/lane) in d_ws.
// Input staging uses global_load_async_to_lds_b128; final codes leave LDS
// via global_store_async_from_lds_b128 (both ASYNCcnt-tracked).
// ---------------------------------------------------------------------------

typedef __attribute__((ext_vector_type(16))) __bf16       v16bf;
typedef __attribute__((ext_vector_type(8)))  float        v8f;
typedef __attribute__((ext_vector_type(4)))  unsigned int u32x4;
typedef __attribute__((ext_vector_type(4)))  float        f32x4;

union Frag { v16bf bf; u32x4 u[2]; };

#define BATCH 8192
#define NCODE 1024
#define KIN   512
#define ROWS  16

// Padded LDS strides (pad 16 B/row -> bank step 4, conflict-free b128)
#define XSTR4   257    // f32x4 per row of x          (1028 f32 = 4112 B)
#define INSTR4  129    // f32x4 per row of inputs     (516 f32  = 2064 B)
#define XBSTRB  2064   // bytes per row of x_bf16     (1032 bf16)
#define EBSTRB  1040   // bytes per row of E_bf16     (520 bf16)

#define X_OFF   0        // f32 [16][1028]  65792 B
#define IN_OFF  65792    // f32 [16][516]   33024 B
#define XB_OFF  98816    // bf16[16][1032]  33024 B
#define EB_OFF  131840   // bf16[16][520]   16640 B
#define SMEM_BYTES 148480 // 145 KB

__device__ __forceinline__ unsigned short f2bf(float f) {
  unsigned int u = __float_as_uint(f);
  u += 0x7FFFu + ((u >> 16) & 1u);   // RNE
  return (unsigned short)(u >> 16);
}
__device__ __forceinline__ unsigned int pack2bf(float lo, float hi) {
  return (unsigned int)f2bf(lo) | ((unsigned int)f2bf(hi) << 16);
}

#define WMMA_BF16(A, B, C) \
  __builtin_amdgcn_wmma_f32_16x16x32_bf16(false, (A), false, (B), (short)0, (C), false, false)

// A-fragment k-position for packing: lane half h (=lane>>4), dword v (0..7)
__device__ __forceinline__ int a_koff(int half, int v) {
  return half * 16 + 2 * v;
}

// ---------------------------------------------------------------------------
// Pack W^T into A-fragment tiles: M = i (32 m-tiles), K = j (32 k-tiles).
// tile = it*32 + kt, 512 bf16/tile (32 lanes x 32 B).
// ---------------------------------------------------------------------------
__global__ __launch_bounds__(256) void prep_wta(const float* __restrict__ W,
                                                unsigned short* __restrict__ Wta) {
  int id   = blockIdx.x * 256 + threadIdx.x;   // 0..262143, 2 bf16 each
  int v    = id & 7;
  int lane = (id >> 3) & 31;
  int it   = (id >> 8) & 31;
  int kt   = id >> 13;
  int i  = it * 16 + (lane & 15);
  int k0 = kt * 32 + a_koff(lane >> 4, v);     // K = j
  int dst = (it * 32 + kt) * 512 + lane * 16 + 2 * v;
  Wta[dst]     = f2bf(W[k0 * KIN + i]);        // W^T[i][j] = W[j][i]
  Wta[dst + 1] = f2bf(W[(k0 + 1) * KIN + i]);
}

// Pack W into A-fragment tiles: M = j (64 m-tiles), K = i (16 k-tiles).
__global__ __launch_bounds__(256) void prep_wa(const float* __restrict__ W,
                                               unsigned short* __restrict__ Wa) {
  int id   = blockIdx.x * 256 + threadIdx.x;   // 0..262143
  int v    = id & 7;
  int lane = (id >> 3) & 31;
  int jt   = (id >> 8) & 63;
  int kt   = id >> 14;
  int j  = jt * 16 + (lane & 15);
  int k0 = kt * 32 + a_koff(lane >> 4, v);
  int dst = (jt * 16 + kt) * 512 + lane * 16 + 2 * v;
  Wa[dst]     = f2bf(W[j * KIN + k0]);
  Wa[dst + 1] = f2bf(W[j * KIN + k0 + 1]);
}

// ---------------------------------------------------------------------------
// Main fused solver: 256 threads = 8 waves, grid = 8192/16 = 512.
// ---------------------------------------------------------------------------
__global__ __launch_bounds__(256) void sc_iterate(const float* __restrict__ inp,
                                                  const unsigned short* __restrict__ Wta,
                                                  const unsigned short* __restrict__ Wa,
                                                  float* __restrict__ out) {
  extern __shared__ char smem[];
  f32x4* s_x4  = (f32x4*)(smem + X_OFF);
  f32x4* s_in4 = (f32x4*)(smem + IN_OFF);
  char*  s_xb  = smem + XB_OFF;
  char*  s_eb  = smem + EB_OFF;

  const int tid  = threadIdx.x;
  const int w    = tid >> 5;
  const int lane = tid & 31;
  const int nl   = lane & 15;     // batch row owned by this lane (N index)
  const int hi   = lane >> 4;
  const size_t rowBase = (size_t)blockIdx.x * ROWS;

  auto loadA = [&](const unsigned short* base, int tile) -> v16bf {
    const char* p = (const char*)base + tile * 1024 + lane * 32;
    Frag f;
    f.u[0] = *(const u32x4*)(p);
    f.u[1] = *(const u32x4*)(p + 16);
    return f.bf;
  };
  auto loadB = [&](const char* base, int rowStrideB, int ktByte) -> v16bf {
    const char* p = base + nl * rowStrideB + ktByte + hi * 32;
    Frag f;
    f.u[0] = *(const u32x4*)(p);
    f.u[1] = *(const u32x4*)(p + 16);
    return f.bf;
  };

  // Lane-private RMSprop state: 8 tiles x 8 consecutive j per lane, in VGPRs.
  float a1r[64], a2r[64];
#pragma unroll
  for (int i = 0; i < 64; ++i) { a1r[i] = 0.f; a2r[i] = 0.f; }

  // ---- init: zero x + bf16 shadow; async-stage inputs into LDS ----
  {
    const f32x4 z4 = {0.f, 0.f, 0.f, 0.f};
    for (int i = tid; i < ROWS * XSTR4; i += 256) s_x4[i] = z4;
    const u32x4 zu = {0u, 0u, 0u, 0u};
    u32x4* xb4 = (u32x4*)s_xb;
    for (int i = tid; i < ROWS * XBSTRB / 16; i += 256) xb4[i] = zu;

    for (int i = tid; i < ROWS * KIN / 4; i += 256) {
      int row = i >> 7, col = i & 127;                    // 128 f32x4 per row
      unsigned int ldsb = (unsigned int)(size_t)(smem + IN_OFF) +
                          (unsigned int)(row * (INSTR4 * 16) + col * 16);
      unsigned long long ga = (unsigned long long)(const void*)
                              (inp + (rowBase + row) * KIN + col * 4);
      asm volatile("global_load_async_to_lds_b128 %0, %1, off"
                   :: "v"(ldsb), "v"(ga) : "memory");
    }
    asm volatile("s_wait_asynccnt 0x0" ::: "memory");
  }
  __syncthreads();

  const v8f Z8 = {0.f, 0.f, 0.f, 0.f, 0.f, 0.f, 0.f, 0.f};

  for (int s = 0; s < 10; ++s) {
    // ---- GEMM1': E^T = W^T * x^T.  Wave owns 4 of 32 i-tiles. ----
    v8f acc[4];
#pragma unroll
    for (int t = 0; t < 4; ++t) acc[t] = Z8;
    for (int kt = 0; kt < 32; ++kt) {
      v16bf b = loadB(s_xb, XBSTRB, kt * 64);             // x^T fragment
#pragma unroll
      for (int t = 0; t < 4; ++t)
        acc[t] = WMMA_BF16(loadA(Wta, (w * 4 + t) * 32 + kt), b, acc[t]);
    }
    // E = acc - inputs  (lane: batch row nl, i = it*16 + hi*8 + r)
#pragma unroll
    for (int t = 0; t < 4; ++t) {
      int it = w * 4 + t;
      int inIdx = nl * INSTR4 + it * 4 + hi * 2;          // f32x4 index
      f32x4 i0 = s_in4[inIdx], i1 = s_in4[inIdx + 1];
      u32x4 e;
      e.x = pack2bf(acc[t][0] - i0[0], acc[t][1] - i0[1]);
      e.y = pack2bf(acc[t][2] - i0[2], acc[t][3] - i0[3]);
      e.z = pack2bf(acc[t][4] - i1[0], acc[t][5] - i1[1]);
      e.w = pack2bf(acc[t][6] - i1[2], acc[t][7] - i1[3]);
      *(u32x4*)(s_eb + nl * EBSTRB + it * 32 + hi * 16) = e;
    }
    // warm caches with GEMM2' first A-tiles while waiting at the barrier
#pragma unroll
    for (int t = 0; t < 4; ++t)
      __builtin_prefetch((const char*)Wa + ((w * 8 + t) * 16) * 1024 + lane * 32, 0, 3);
    __syncthreads();

    // ---- GEMM2': G^T = W * E^T.  Wave owns 8 of 64 j-tiles. ----
    v8f acc2[8];
#pragma unroll
    for (int t = 0; t < 8; ++t) acc2[t] = Z8;
    for (int kt = 0; kt < 16; ++kt) {
      v16bf b = loadB(s_eb, EBSTRB, kt * 64);             // E^T fragment
#pragma unroll
      for (int t = 0; t < 8; ++t)
        acc2[t] = WMMA_BF16(loadA(Wa, (w * 8 + t) * 16 + kt), b, acc2[t]);
    }

    // ---- RMSprop + momentum: lane owns batch row nl, 8 consecutive j ----
#pragma unroll
    for (int t = 0; t < 8; ++t) {
      int jt = w * 8 + t;
      int base = nl * XSTR4 + jt * 4 + hi * 2;            // f32x4 index
      f32x4 x0 = s_x4[base], x1 = s_x4[base + 1];
      float xs[8];
#pragma unroll
      for (int r = 0; r < 8; ++r) {
        float x  = (r < 4) ? x0[r] : x1[r - 4];
        float aa = a1r[t * 8 + r];
        float bb = a2r[t * 8 + r];
        // g = 2*(xW - in)W^T + (GAMMA + GAMMA/10) * x / sqrt(x^2 + 1e-6)
        float g = 2.0f * acc2[t][r] +
                  0.11f * x * __builtin_amdgcn_rsqf(x * x + 1e-6f);
        aa = 0.9f * aa + 0.1f * g * g;
        float upd = 0.009f * g * __builtin_amdgcn_rsqf(aa + 1e-8f);
        bb = 0.9f * bb - upd;
        x  = x + 0.9f * bb - upd;
        xs[r] = x;
        a1r[t * 8 + r] = aa;
        a2r[t * 8 + r] = bb;
      }
      f32x4 ox0 = {xs[0], xs[1], xs[2], xs[3]}, ox1 = {xs[4], xs[5], xs[6], xs[7]};
      s_x4[base] = ox0; s_x4[base + 1] = ox1;
      u32x4 xb;
      xb.x = pack2bf(xs[0], xs[1]); xb.y = pack2bf(xs[2], xs[3]);
      xb.z = pack2bf(xs[4], xs[5]); xb.w = pack2bf(xs[6], xs[7]);
      *(u32x4*)(s_xb + nl * XBSTRB + jt * 32 + hi * 16) = xb;
    }
    // warm caches with GEMM1' first A-tiles while waiting at the barrier
#pragma unroll
    for (int t = 0; t < 4; ++t)
      __builtin_prefetch((const char*)Wta + ((w * 4 + t) * 32) * 1024 + lane * 32, 0, 3);
    __syncthreads();
  }

  // ---- final codes: LDS -> global via async store-from-LDS (de-pad rows) ----
  for (int i = tid; i < ROWS * NCODE / 4; i += 256) {
    int row = i >> 8, col = i & 255;                      // 256 f32x4 per row
    unsigned int ldsb = (unsigned int)(size_t)(smem + X_OFF) +
                        (unsigned int)(row * (XSTR4 * 16) + col * 16);
    unsigned long long ga = (unsigned long long)(void*)
                            (out + (rowBase + row) * NCODE + col * 4);
    asm volatile("global_store_async_from_lds_b128 %0, %1, off"
                 :: "v"(ga), "v"(ldsb) : "memory");
  }
  asm volatile("s_wait_asynccnt 0x0" ::: "memory");
}

// ---------------------------------------------------------------------------
extern "C" void kernel_launch(void* const* d_in, const int* in_sizes, int n_in,
                              void* d_out, int out_size, void* d_ws, size_t ws_size,
                              hipStream_t stream) {
  (void)in_sizes; (void)n_in; (void)out_size; (void)ws_size;
  const float* inp = (const float*)d_in[0];   // [8192, 512]
  const float* W   = (const float*)d_in[1];   // [1024, 512]
  unsigned short* Wta = (unsigned short*)d_ws;        // 1 MB packed W^T (GEMM1' A)
  unsigned short* Wa  = Wta + 512 * 1024;             // 1 MB packed W   (GEMM2' A)

  prep_wta<<<1024, 256, 0, stream>>>(W, Wta);
  prep_wa <<<1024, 256, 0, stream>>>(W, Wa);
  sc_iterate<<<BATCH / ROWS, 256, SMEM_BYTES, stream>>>(inp, Wta, Wa, (float*)d_out);
}
